// BotRGCN_14224931684700
// MI455X (gfx1250) — matrix operand order repaired and checked by reading, not compiled, not run on hardware
//
#include <hip/hip_runtime.h>
#include <hip/hip_bf16.h>

typedef __attribute__((ext_vector_type(16))) __bf16 v16bf;
typedef __attribute__((ext_vector_type(8)))  __bf16 v8bf;
typedef __attribute__((ext_vector_type(8)))  float  v8f;

#define XCOLS 1553   // 6 + 768 + 11 + 768
#define HDIM  128
#define RELS  2

__device__ __forceinline__ __bf16 f2bf(float f) {
  union { float f; unsigned u; } a; a.f = f;
  unsigned r = a.u + 0x7FFFu + ((a.u >> 16) & 1u);   // round-to-nearest-even
  union { unsigned short s; __bf16 b; } o; o.s = (unsigned short)(r >> 16);
  return o.b;
}

// ---------------------------------------------------------------------------
// Repack a K x N (row-major, f32) weight matrix into WMMA B-fragment order:
// tile t = nt*KT + kt ; per tile 32 lanes x 16 bf16 (32B contiguous per lane).
// Dense 16-bit B 32x16 layout: lanes 0-15 hold K=kt*32+0..7,16..23 (N = lane),
// lanes 16-31 hold K=kt*32+8..15,24..31 (N = lane-16).
// ---------------------------------------------------------------------------
__global__ void pack_b_kernel(const float* __restrict__ B, __bf16* __restrict__ P,
                              int KT, int NT) {
  int idx = blockIdx.x * blockDim.x + threadIdx.x;
  int total = KT * NT * 512;
  if (idx >= total) return;
  int i    = idx & 15;
  int lane = (idx >> 4) & 31;
  int t    = idx >> 9;
  int kt   = t % KT;
  int nt   = t / KT;
  int n = nt * 16 + (lane & 15);
  int k = kt * 32 + ((lane >= 16) ? 8 : 0) + ((i < 8) ? i : i + 8);
  P[idx] = f2bf(B[(size_t)k * (NT * 16) + n]);
}

// ---------------------------------------------------------------------------
// WMMA GEMM: C[M x Ncols] = act(A[M x K] * B + bias)
// Each wave: one 16-row tile x 128 cols (8 WMMA n-tiles); blockIdx.y covers
// additional 128-col slabs. ACT: 0=none, 1=leaky(0.01), 2=prelu(per-col a).
// AF32: A is f32 (convert on the fly) else bf16.
// 16-bit A 16x32 layout: lane<16 -> K 0..7 & 16..23 ; lane>=16 -> 8..15 & 24..31.
// ---------------------------------------------------------------------------
template<int ACT, bool AF32>
__global__ void gemm_wmma(const void* __restrict__ Av, long lda,
                          const __bf16* __restrict__ Bpack, int KT,
                          const float* __restrict__ bias,
                          const float* __restrict__ prelu,
                          float* __restrict__ outF, long ldof,
                          __bf16* __restrict__ outB, long ldob,
                          int Mtiles) {
  const int lane  = threadIdx.x & 31;
  const int wave  = threadIdx.x >> 5;
  const int tileM = blockIdx.x * 8 + wave;
  if (tileM >= Mtiles) return;               // whole-wave uniform exit (EXEC all-1 for WMMA)
  const bool hi     = lane >= 16;
  const int  lanelo = lane & 15;
  const int  m      = tileM * 16 + lanelo;
  const int  ntBase = blockIdx.y * 8;

  v8f c[8] = {};
  const v16bf* Bt = (const v16bf*)Bpack;

  for (int kt = 0; kt < KT; ++kt) {
    v16bf a;
    const int kb = kt * 32 + (hi ? 8 : 0);
    if (AF32) {
      const float* Ar = (const float*)Av + (size_t)m * lda + kb;
      #pragma unroll
      for (int i = 0; i < 8; ++i) a[i] = f2bf(Ar[i]);
      #pragma unroll
      for (int i = 0; i < 8; ++i) a[8 + i] = f2bf(Ar[16 + i]);
    } else {
      const __bf16* Ar = (const __bf16*)Av + (size_t)m * lda + kb;
      v8bf lo = *(const v8bf*)(Ar);
      v8bf h8 = *(const v8bf*)(Ar + 16);
      #pragma unroll
      for (int i = 0; i < 8; ++i) { a[i] = lo[i]; a[8 + i] = h8[i]; }
    }
    #pragma unroll
    for (int j = 0; j < 8; ++j) {
      v16bf b = Bt[(size_t)((ntBase + j) * KT + kt) * 32 + lane];
      c[j] = __builtin_amdgcn_wmma_f32_16x16x32_bf16(
                 false, a, false, b, (short)0, c[j], false, false);
    }
  }

  // C/D layout: VGPR e -> (lanes 0-15: M=e, N=lane) (lanes 16-31: M=8+e, N=lane-16)
  #pragma unroll
  for (int j = 0; j < 8; ++j) {
    const int   n  = (ntBase + j) * 16 + lanelo;
    const float bs = bias[n];
    float pa = 0.f;
    if (ACT == 2) pa = prelu[n];
    #pragma unroll
    for (int e = 0; e < 8; ++e) {
      const int mr = tileM * 16 + e + (hi ? 8 : 0);
      float v = c[j][e] + bs;
      if (ACT == 1) v = (v >= 0.f) ? v : 0.01f * v;
      if (ACT == 2) v = (v >= 0.f) ? v : pa * v;
      if (outF) outF[(size_t)mr * ldof + n] = v;
      if (outB) outB[(size_t)mr * ldob + n] = f2bf(v);
    }
  }
}

// Tiny-K projections (num: K=6, cat: K=11) -> h512 cols [256,384) and [384,512)
__global__ void smallproj_kernel(const float* __restrict__ x,
                                 const float* __restrict__ Wn, const float* __restrict__ bn,
                                 const float* __restrict__ Wc, const float* __restrict__ bc,
                                 __bf16* __restrict__ h512) {
  const int row = blockIdx.x;
  const int col = threadIdx.x;          // 128 threads
  const float* xr = x + (size_t)row * XCOLS;
  float sn = bn[col];
  #pragma unroll
  for (int k = 0; k < 6; ++k) sn += xr[k] * Wn[k * HDIM + col];
  float sc = bc[col];
  #pragma unroll
  for (int k = 0; k < 11; ++k) sc += xr[774 + k] * Wc[k * HDIM + col];
  sn = (sn >= 0.f) ? sn : 0.01f * sn;
  sc = (sc >= 0.f) ? sc : 0.01f * sc;
  __bf16* hr = h512 + (size_t)row * 512;
  hr[256 + col] = f2bf(sn);
  hr[384 + col] = f2bf(sc);
}

__global__ void zero_i32_kernel(int* p, long n) {
  long i = (long)blockIdx.x * blockDim.x + threadIdx.x;
  if (i < n) p[i] = 0;
}

__global__ void cnt_kernel(const int* __restrict__ dst, const int* __restrict__ et,
                           int* __restrict__ cnt, int E) {
  int e = blockIdx.x * blockDim.x + threadIdx.x;
  if (e >= E) return;
  atomicAdd(&cnt[(size_t)dst[e] * RELS + et[e]], 1);
}

__global__ void inv_kernel(const int* __restrict__ cnt, float* __restrict__ inv, long n) {
  long i = (long)blockIdx.x * blockDim.x + threadIdx.x;
  if (i >= n) return;
  int c = cnt[i];
  inv[i] = 1.0f / (float)(c > 1 ? c : 1);
}

// base_rel layout per row (stride 384): [out(128) | hrel0(128) | hrel1(128)]
// One wave per edge: out[dst] += hrel[r][src] * invcnt[dst,r]   (in-place; reads
// cols >=128, writes cols <128 -> no aliasing hazard).
__global__ void scatter_kernel(const int* __restrict__ src, const int* __restrict__ dst,
                               const int* __restrict__ et, const float* __restrict__ invc,
                               float* __restrict__ base_rel, int E) {
  int g    = blockIdx.x * blockDim.x + threadIdx.x;
  int e    = g >> 5;
  int lane = g & 31;
  if (e >= E) return;
  int s = src[e], d = dst[e], r = et[e];
  float w = invc[(size_t)d * RELS + r];
  const float* hs = base_rel + (size_t)s * 384 + 128 + (size_t)r * 128 + lane * 4;
  float*       od = base_rel + (size_t)d * 384 + lane * 4;
  float4 hv = *(const float4*)hs;
  atomicAdd(od + 0, hv.x * w);
  atomicAdd(od + 1, hv.y * w);
  atomicAdd(od + 2, hv.z * w);
  atomicAdd(od + 3, hv.w * w);
}

// base_rel[:,0:128] (stride 384, f32) -> h_bf16 (stride 128)
__global__ void tobf_kernel(const float* __restrict__ in, __bf16* __restrict__ out, long n) {
  long i = (long)blockIdx.x * blockDim.x + threadIdx.x;
  if (i >= n) return;
  long row = i >> 7;
  int  col = (int)(i & 127);
  out[i] = f2bf(in[row * 384 + col]);
}

__global__ void biasbig_kernel(const float* __restrict__ b, float* __restrict__ bb) {
  int i = blockIdx.x * blockDim.x + threadIdx.x;
  if (i < 384) bb[i] = (i < 128) ? b[i] : 0.f;
}

extern "C" void kernel_launch(void* const* d_in, const int* in_sizes, int n_in,
                              void* d_out, int out_size, void* d_ws, size_t ws_size,
                              hipStream_t stream) {
  const float* x     = (const float*)d_in[0];
  const int*   eidx  = (const int*)d_in[1];
  const int*   etype = (const int*)d_in[2];
  const float* W_des = (const float*)d_in[3];  const float* b_des = (const float*)d_in[4];
  const float* W_tw  = (const float*)d_in[5];  const float* b_tw  = (const float*)d_in[6];
  const float* W_num = (const float*)d_in[7];  const float* b_num = (const float*)d_in[8];
  const float* W_cat = (const float*)d_in[9];  const float* b_cat = (const float*)d_in[10];
  const float* W_in  = (const float*)d_in[11]; const float* b_in  = (const float*)d_in[12];
  const float* prelu = (const float*)d_in[13];
  const float* root1 = (const float*)d_in[14]; const float* rel1  = (const float*)d_in[15];
  const float* bias1 = (const float*)d_in[16];
  const float* root2 = (const float*)d_in[17]; const float* rel2  = (const float*)d_in[18];
  const float* bias2 = (const float*)d_in[19];
  const float* W_cls = (const float*)d_in[20]; const float* b_cls = (const float*)d_in[21];

  const int N = in_sizes[0] / XCOLS;          // 100000
  const int E = in_sizes[2];                  // 1600000
  const int* src = eidx;
  const int* dst = eidx + E;

  // -------- workspace layout (bytes) --------
  char* ws = (char*)d_ws;
  size_t off = 0;
  auto carve = [&](size_t bytes) { size_t o = off; off += (bytes + 255) & ~(size_t)255; return o; };
  __bf16* h512    = (__bf16*)(ws + carve((size_t)N * 512 * 2));   // concat features, bf16
  __bf16* h_bf    = (__bf16*)(ws + carve((size_t)N * 128 * 2));   // current h, bf16
  float*  base    = (float*)(ws + carve((size_t)N * 384 * 4));    // [out|hrel0|hrel1]
  int*    cntI    = (int*)(ws + carve((size_t)N * RELS * 4));
  float*  invc    = (float*)(ws + carve((size_t)N * RELS * 4));
  __bf16* Pdes    = (__bf16*)(ws + carve((size_t)24 * 8 * 512 * 2));
  __bf16* Ptw     = (__bf16*)(ws + carve((size_t)24 * 8 * 512 * 2));
  __bf16* Pin     = (__bf16*)(ws + carve((size_t)16 * 8 * 512 * 2));
  __bf16* PB1     = (__bf16*)(ws + carve((size_t)4 * 24 * 512 * 2));
  __bf16* PB2     = (__bf16*)(ws + carve((size_t)4 * 24 * 512 * 2));
  __bf16* Pcls    = (__bf16*)(ws + carve((size_t)4 * 8 * 512 * 2));
  float*  bb1     = (float*)(ws + carve(384 * 4));
  float*  bb2     = (float*)(ws + carve(384 * 4));

  const int Mtiles = N / 16;                  // 6250 (N divisible by 16)
  const dim3 blk(256);
  const int gemmGX = (Mtiles + 7) / 8;

  // -------- weight repack (bf16, WMMA B-fragment order) --------
  auto packLaunch = [&](const float* B, __bf16* P, int KT, int NT) {
    int total = KT * NT * 512;
    pack_b_kernel<<<(total + 255) / 256, 256, 0, stream>>>(B, P, KT, NT);
  };
  packLaunch(W_des, Pdes, 24, 8);
  packLaunch(W_tw,  Ptw,  24, 8);
  packLaunch(W_in,  Pin,  16, 8);
  packLaunch(root1, PB1,                 4, 8);   // nt 0..7
  packLaunch(rel1,  PB1 + (size_t)8  * 4 * 512, 4, 8);  // nt 8..15
  packLaunch(rel1 + 128 * 128, PB1 + (size_t)16 * 4 * 512, 4, 8); // nt 16..23
  packLaunch(root2, PB2,                 4, 8);
  packLaunch(rel2,  PB2 + (size_t)8  * 4 * 512, 4, 8);
  packLaunch(rel2 + 128 * 128, PB2 + (size_t)16 * 4 * 512, 4, 8);
  packLaunch(W_cls, Pcls, 4, 8);
  biasbig_kernel<<<2, 192, 0, stream>>>(bias1, bb1);
  biasbig_kernel<<<2, 192, 0, stream>>>(bias2, bb2);

  // -------- per-(dst, relation) edge counts -> 1/max(cnt,1) (shared by both layers)
  long nc = (long)N * RELS;
  zero_i32_kernel<<<(nc + 255) / 256, 256, 0, stream>>>(cntI, nc);
  cnt_kernel<<<(E + 255) / 256, 256, 0, stream>>>(dst, etype, cntI, E);
  inv_kernel<<<(nc + 255) / 256, 256, 0, stream>>>(cntI, invc, nc);

  // -------- feature projections -> h512 (bf16) --------
  smallproj_kernel<<<N, 128, 0, stream>>>(x, W_num, b_num, W_cat, b_cat, h512);
  // des: x[:,785:1553] @ W_des, leaky -> h512[:,0:128]
  gemm_wmma<1, true><<<dim3(gemmGX, 1), blk, 0, stream>>>(
      (const void*)(x + 785), XCOLS, Pdes, 24, b_des, nullptr,
      nullptr, 0, h512, 512, Mtiles);
  // tweet: x[:,6:774] @ W_tweet, leaky -> h512[:,128:256]
  gemm_wmma<1, true><<<dim3(gemmGX, 1), blk, 0, stream>>>(
      (const void*)(x + 6), XCOLS, Ptw, 24, b_tw, nullptr,
      nullptr, 0, h512 + 128, 512, Mtiles);

  // -------- input layer: h = prelu(h512 @ W_in + b_in) -> h_bf --------
  gemm_wmma<2, false><<<dim3(gemmGX, 1), blk, 0, stream>>>(
      (const void*)h512, 512, Pin, 16, b_in, prelu,
      nullptr, 0, h_bf, 128, Mtiles);

  // -------- RGCN layer 1 --------
  gemm_wmma<0, false><<<dim3(gemmGX, 3), blk, 0, stream>>>(
      (const void*)h_bf, 128, PB1, 4, bb1, nullptr,
      base, 384, nullptr, 0, Mtiles);
  scatter_kernel<<<(E * 32 + 255) / 256, 256, 0, stream>>>(src, dst, etype, invc, base, E);
  tobf_kernel<<<((long)N * 128 + 255) / 256, 256, 0, stream>>>(base, h_bf, (long)N * 128);

  // -------- RGCN layer 2 --------
  gemm_wmma<0, false><<<dim3(gemmGX, 3), blk, 0, stream>>>(
      (const void*)h_bf, 128, PB2, 4, bb2, nullptr,
      base, 384, nullptr, 0, Mtiles);
  scatter_kernel<<<(E * 32 + 255) / 256, 256, 0, stream>>>(src, dst, etype, invc, base, E);
  tobf_kernel<<<((long)N * 128 + 255) / 256, 256, 0, stream>>>(base, h_bf, (long)N * 128);

  // -------- classifier: out = h @ W_cls + b_cls --------
  gemm_wmma<0, false><<<dim3(gemmGX, 1), blk, 0, stream>>>(
      (const void*)h_bf, 128, Pcls, 4, b_cls, nullptr,
      (float*)d_out, 128, nullptr, 0, Mtiles);
}
